// LSTMCell_47863115547325
// MI455X (gfx1250) — matrix operand order repaired and checked by reading, not compiled, not run on hardware
//
#include <hip/hip_runtime.h>
#include <stdint.h>

typedef uint32_t u32;
typedef __attribute__((ext_vector_type(4)))  u32    v4u;
typedef __attribute__((ext_vector_type(2)))  __bf16 v2bf;
typedef __attribute__((ext_vector_type(16))) __bf16 v16bf;
typedef __attribute__((ext_vector_type(8)))  float  v8f;

#define B_ROWS 16384
#define DD 512
#define UU 512
#define NT 32          // K = 1024, TK = 32

static __device__ __forceinline__ u32 pack2(float lo, float hi) {
#if __has_builtin(__builtin_amdgcn_cvt_pk_bf16_f32)
  v2bf r = __builtin_amdgcn_cvt_pk_bf16_f32(lo, hi);
  return __builtin_bit_cast(u32, r);
#else
  v2bf p;
  p.x = (__bf16)lo;
  p.y = (__bf16)hi;
  return __builtin_bit_cast(u32, p);
#endif
}

struct Q2 { v4u a, b; };
static __device__ __forceinline__ v16bf make_frag(v4u lo, v4u hi) {
  Q2 q; q.a = lo; q.b = hi;
  return __builtin_bit_cast(v16bf, q);
}
static __device__ __forceinline__ float sigm(float x) {
  return 1.0f / (1.0f + __expf(-x));
}
static __device__ __forceinline__ float tanh_(float x) {
  return 2.0f * sigm(2.0f * x) - 1.0f;
}

// WG tile: 128 rows x 256 gate-cols (u-tile = 64). 8 waves as 2(M) x 4(N),
// each wave: 64x64 -> 4 A-frags x 4 B-frags -> 16 WMMAs, 16 ds_load_b128.
// Gate interleave in tile-column space: c -> gate g=(c>>4)&3,
//                                       u  -> u0 + (c>>6)*16 + (c&15)
// so wave waveN (cols waveN*64 + ci*16) holds gates ci=0..3 at u-block
// u0 + waveN*16 : the LSTM epilogue stays wave-local.
__global__ __launch_bounds__(256) void lstm_cell_wmma(
    const float* __restrict__ X,   // [B, 512]
    const float* __restrict__ Hp,  // [B, 512]
    const float* __restrict__ Cp,  // [B, 512]
    const float* __restrict__ W,   // [1024, 2048]
    const float* __restrict__ bias,// [2048]
    float* __restrict__ outH, float* __restrict__ outC) {
  __shared__ v4u ldsA[2][128 * 5];  // 128 rows x 32 k bf16, stride 80B
  __shared__ v4u ldsB[2][256 * 5];  // 256 c   x 32 k bf16 (transposed), stride 80B

  const int tid  = threadIdx.x;
  const int lane = tid & 31;
  const int wave = tid >> 5;
  const int rowTile = blockIdx.y * 128;
  const int u0      = blockIdx.x * 64;

  // ---- fill roles ----
  const int aRow  = tid >> 1;        // 0..127
  const int aHalf = tid & 1;
  const int bKP   = tid & 15;        // k-pair (rows 2k, 2k+1)
  const int bOct  = tid >> 4;        // two 8-col groups: bOct, bOct+16
  const int cBase0 = bOct * 8;            // 0..120
  const int cBase1 = (bOct + 16) * 8;     // 128..248

  auto wcol = [&](int cb) -> long {   // tile col -> column in W
    const int g    = (cb >> 4) & 3;
    const int uoff = (cb >> 6) * 16 + (cb & 15);
    return (long)g * UU + u0 + uoff;
  };

  const float* aRowX = X  + (size_t)(rowTile + aRow) * DD + aHalf * 16;
  const float* aRowH = Hp + (size_t)(rowTile + aRow) * UU + aHalf * 16;
  const float* bCol0 = W + wcol(cBase0) + (size_t)(bKP * 2) * (4 * UU);
  const float* bCol1 = W + wcol(cBase1) + (size_t)(bKP * 2) * (4 * UU);

  float4 aR[4];
  float4 bR[8];  // [group0: row2k(2), row2k+1(2)] [group1: ...]

  auto gload = [&](int kt) {
    const float* asrc = (kt < 16) ? (aRowX + kt * 32) : (aRowH + (kt - 16) * 32);
    const float4* a4 = (const float4*)asrc;
    aR[0] = a4[0]; aR[1] = a4[1]; aR[2] = a4[2]; aR[3] = a4[3];

    const size_t koff = (size_t)(kt * 32) * (4 * UU);
    const float4* g0  = (const float4*)(bCol0 + koff);
    const float4* g0n = (const float4*)(bCol0 + koff + 4 * UU);
    bR[0] = g0[0]; bR[1] = g0[1]; bR[2] = g0n[0]; bR[3] = g0n[1];
    const float4* g1  = (const float4*)(bCol1 + koff);
    const float4* g1n = (const float4*)(bCol1 + koff + 4 * UU);
    bR[4] = g1[0]; bR[5] = g1[1]; bR[6] = g1n[0]; bR[7] = g1n[1];
  };

  auto lstore = [&](int buf) {
    // A: 16 f32 -> 16 bf16 contiguous in k
    u32 p0 = pack2(aR[0].x, aR[0].y), p1 = pack2(aR[0].z, aR[0].w);
    u32 p2 = pack2(aR[1].x, aR[1].y), p3 = pack2(aR[1].z, aR[1].w);
    u32 p4 = pack2(aR[2].x, aR[2].y), p5 = pack2(aR[2].z, aR[2].w);
    u32 p6 = pack2(aR[3].x, aR[3].y), p7 = pack2(aR[3].z, aR[3].w);
    char* ab = (char*)&ldsA[buf][0] + aRow * 80 + aHalf * 32;
    *(v4u*)(ab)      = (v4u){p0, p1, p2, p3};
    *(v4u*)(ab + 16) = (v4u){p4, p5, p6, p7};
    // B: transpose-scatter, pack k-pair per column
    char* bb = (char*)&ldsB[buf][0] + bKP * 4;
    const float* ra0 = (const float*)&bR[0];
    const float* rb0 = (const float*)&bR[2];
    const float* ra1 = (const float*)&bR[4];
    const float* rb1 = (const float*)&bR[6];
#pragma unroll
    for (int j = 0; j < 8; ++j)
      *(u32*)(bb + (cBase0 + j) * 80) = pack2(ra0[j], rb0[j]);
#pragma unroll
    for (int j = 0; j < 8; ++j)
      *(u32*)(bb + (cBase1 + j) * 80) = pack2(ra1[j], rb1[j]);
  };

  // ---- compute roles ----
  const int r15 = lane & 15;
  const int hh  = lane >> 4;
  const int waveM = wave & 1;   // 0..1 : rows waveM*64
  const int waveN = wave >> 1;  // 0..3 : cols waveN*64

  v8f acc[16];
  v8f zero = {0.f, 0.f, 0.f, 0.f, 0.f, 0.f, 0.f, 0.f};
#pragma unroll
  for (int i = 0; i < 16; ++i) acc[i] = zero;

  auto compute = [&](int buf) {
    // 4 B fragments (gates 0..3 at this wave's u-block), reused across 4 M-strips
    v4u bl[4], bh[4];
#pragma unroll
    for (int ci = 0; ci < 4; ++ci) {
      const char* bb = (const char*)&ldsB[buf][0] +
                       (waveN * 64 + ci * 16 + r15) * 80 + hh * 16;
      bl[ci] = *(const v4u*)bb;
      bh[ci] = *(const v4u*)(bb + 32);
    }
#pragma unroll
    for (int mi = 0; mi < 4; ++mi) {
      const char* ab = (const char*)&ldsA[buf][0] +
                       (waveM * 64 + mi * 16 + r15) * 80 + hh * 16;
      v16bf af = make_frag(*(const v4u*)ab, *(const v4u*)(ab + 32));
#pragma unroll
      for (int ci = 0; ci < 4; ++ci) {
        v16bf bf = make_frag(bl[ci], bh[ci]);
        acc[mi * 4 + ci] = __builtin_amdgcn_wmma_f32_16x16x32_bf16(
            false, af, false, bf, (short)0, acc[mi * 4 + ci], false, false);
      }
    }
  };

  // bias preload: exactly one u per lane
  const int uLane = u0 + waveN * 16 + r15;
  const float bI = bias[uLane];
  const float bF = bias[UU + uLane];
  const float bC = bias[2 * UU + uLane];
  const float bO = bias[3 * UU + uLane];

  gload(0);
  lstore(0);
  __syncthreads();
  for (int kt = 0; kt < NT; ++kt) {
    const int cur = kt & 1;
    if (kt + 1 < NT) gload(kt + 1);   // overlap global latency with WMMA
    compute(cur);
    if (kt + 1 < NT) lstore(cur ^ 1);
    __syncthreads();
  }

  // ---- fused LSTM epilogue (wave-local: acc[mi*4 + gate]) ----
#pragma unroll
  for (int mi = 0; mi < 4; ++mi) {
    const int row0 = rowTile + waveM * 64 + mi * 16 + hh * 8;
    v8f ia = acc[mi * 4 + 0];
    v8f fa = acc[mi * 4 + 1];
    v8f ca = acc[mi * 4 + 2];
    v8f oa = acc[mi * 4 + 3];
#pragma unroll
    for (int r = 0; r < 8; ++r) {
      const size_t idx = (size_t)(row0 + r) * UU + uLane;
      const float cp = Cp[idx];
      const float iv = sigm(ia[r] + bI);
      const float fv = sigm(fa[r] + bF);
      const float cb = tanh_(ca[r] + bC);
      const float ov = sigm(oa[r] + bO);
      const float cn = fv * cp + iv * cb;
      outH[idx] = ov * tanh_(cn);
      outC[idx] = cn;
    }
  }
}

extern "C" void kernel_launch(void* const* d_in, const int* in_sizes, int n_in,
                              void* d_out, int out_size, void* d_ws, size_t ws_size,
                              hipStream_t stream) {
  const float* X    = (const float*)d_in[0];
  const float* Hp   = (const float*)d_in[1];
  const float* Cp   = (const float*)d_in[2];
  const float* W    = (const float*)d_in[3];
  const float* bias = (const float*)d_in[4];
  float* outH = (float*)d_out;
  float* outC = (float*)d_out + (size_t)B_ROWS * UU;

  dim3 grid(UU / 64, B_ROWS / 128);  // 8 x 128 = 1024 workgroups
  lstm_cell_wmma<<<grid, 256, 0, stream>>>(X, Hp, Cp, W, bias, outH, outC);
}
